// GatLayers_41506563948894
// MI455X (gfx1250) — matrix operand order repaired and checked by reading, not compiled, not run on hardware
//
#include <hip/hip_runtime.h>
#include <hip/hip_bf16.h>

typedef __attribute__((ext_vector_type(16))) _Float16 v16h;
typedef __attribute__((ext_vector_type(8)))  float    v8f;

#define N_NODES 100000
#define N_EDGES 1600000
#define DIM     128           // D_IN == HEADS*C
#define HEADS   2
#define CH      64
#define NEG_SLOPE 0.2f

// ---------- helpers: order-preserving float <-> uint for atomicMax ----------
__device__ __forceinline__ unsigned fflip(float f) {
  unsigned u = __float_as_uint(f);
  return (u & 0x80000000u) ? ~u : (u | 0x80000000u);
}
__device__ __forceinline__ float funflip(unsigned u) {
  u = (u & 0x80000000u) ? (u ^ 0x80000000u) : ~u;
  return __uint_as_float(u);
}

// ---------- W[l][k][n] (f32) -> Wt[l][n][k] (f16, transposed) ----------
__global__ __launch_bounds__(256)
void gat_convW(const float* __restrict__ W, _Float16* __restrict__ Wt, int total) {
  int i = blockIdx.x * blockDim.x + threadIdx.x;
  if (i >= total) return;
  int l = i >> 14;            // 128*128 = 16384 per layer
  int r = i & 16383;
  int n = r >> 7, k = r & 127;
  Wt[(l << 14) + n * DIM + k] = (_Float16)W[(l << 14) + k * DIM + n];
}

// ---------- h = x @ W  via v_wmma_f32_16x16x32_f16 ----------
// grid = 6250 blocks (M tiles), 256 threads = 8 waves = 8 N tiles of 16.
__global__ __launch_bounds__(256)
void gat_gemm(const float* __restrict__ x, const _Float16* __restrict__ Wt,
              float* __restrict__ h) {
  const int wave   = threadIdx.x >> 5;
  const int lane   = threadIdx.x & 31;
  const int lane15 = lane & 15;
  const bool hi    = (lane & 16) != 0;
  const int mtile  = blockIdx.x;
  const int ntile  = wave;

  const float*    xr = x  + (size_t)(mtile * 16 + lane15) * DIM; // A row m
  const _Float16* wr = Wt + (size_t)(ntile * 16 + lane15) * DIM; // B col n (k-contig)

  v8f c = {};
#pragma unroll
  for (int kc = 0; kc < 4; ++kc) {
    // A k order used here (contiguous, same as B): lane<16 -> K 0..15; lane>=16 -> K 16..31
    const int ka = kc * 32 + (hi ? 16 : 0);
    float4 f0 = *(const float4*)(xr + ka);
    float4 f1 = *(const float4*)(xr + ka + 4);
    float4 f2 = *(const float4*)(xr + ka + 8);
    float4 f3 = *(const float4*)(xr + ka + 12);
    v16h a;
    a[0]=(_Float16)f0.x; a[1]=(_Float16)f0.y; a[2]=(_Float16)f0.z; a[3]=(_Float16)f0.w;
    a[4]=(_Float16)f1.x; a[5]=(_Float16)f1.y; a[6]=(_Float16)f1.z; a[7]=(_Float16)f1.w;
    a[8]=(_Float16)f2.x; a[9]=(_Float16)f2.y; a[10]=(_Float16)f2.z; a[11]=(_Float16)f2.w;
    a[12]=(_Float16)f3.x; a[13]=(_Float16)f3.y; a[14]=(_Float16)f3.z; a[15]=(_Float16)f3.w;
    // B uses the same contiguous k order: lanes 0-15 hold K=0..15, lanes 16-31 hold K=16..31
    v16h b = *(const v16h*)(wr + kc * 32 + (hi ? 16 : 0));
    c = __builtin_amdgcn_wmma_f32_16x16x32_f16(false, a, false, b,
                                               (short)0, c, false, false);
  }
  // D 16x16 f32: lanes 0-15 -> M = r, lanes 16-31 -> M = 8+r; N = lane&15
  float* hr = h + (size_t)(mtile * 16 + (hi ? 8 : 0)) * DIM + ntile * 16 + lane15;
#pragma unroll
  for (int r = 0; r < 8; ++r) hr[(size_t)r * DIM] = c[r];
}

// ---------- alpha_s[n][hd], alpha_d[n][hd] ----------
__global__ __launch_bounds__(256)
void gat_alpha(const float* __restrict__ h, const float* __restrict__ a_src,
               const float* __restrict__ a_dst, float* __restrict__ as,
               float* __restrict__ ad, int N) {
  int n = blockIdx.x * blockDim.x + threadIdx.x;
  if (n >= N) return;
  const float* hr = h + (size_t)n * DIM;
#pragma unroll
  for (int hd = 0; hd < HEADS; ++hd) {
    float s = 0.f, d = 0.f;
#pragma unroll 4
    for (int c = 0; c < CH; c += 4) {
      float4 hv = *(const float4*)(hr + hd * CH + c);
      float4 sv = *(const float4*)(a_src + hd * CH + c);
      float4 dv = *(const float4*)(a_dst + hd * CH + c);
      s += hv.x * sv.x + hv.y * sv.y + hv.z * sv.z + hv.w * sv.w;
      d += hv.x * dv.x + hv.y * dv.y + hv.z * dv.z + hv.w * dv.w;
    }
    as[n * HEADS + hd] = s;
    ad[n * HEADS + hd] = d;
  }
}

// ---------- init: xo = bias (+ residual), emax = 0 (below flip(-inf)), denom = 0 ----------
__global__ __launch_bounds__(256)
void gat_init(float* __restrict__ xo, const float* __restrict__ xin,
              const float* __restrict__ bias, unsigned* __restrict__ emax,
              float* __restrict__ denom, int total) {
  int i = blockIdx.x * blockDim.x + threadIdx.x;
  if (i >= total) return;
  int c = i & (DIM - 1), n = i >> 7;
  xo[i] = bias[c] + (xin ? xin[i] : 0.0f);
  if (c < HEADS) { emax[n * HEADS + c] = 0u; denom[n * HEADS + c] = 0.0f; }
}

// ---------- edge pass 1: segment max ----------
__global__ __launch_bounds__(256)
void gat_edge_max(const int* __restrict__ src, const int* __restrict__ dst,
                  const float* __restrict__ as, const float* __restrict__ ad,
                  unsigned* __restrict__ emax, int E) {
  int e = blockIdx.x * blockDim.x + threadIdx.x;
  if (e >= E) return;
  int s = src[e], d = dst[e];
#pragma unroll
  for (int hd = 0; hd < HEADS; ++hd) {
    float v = as[s * HEADS + hd] + ad[d * HEADS + hd];
    v = v > 0.0f ? v : NEG_SLOPE * v;
    atomicMax(&emax[d * HEADS + hd], fflip(v));
  }
}

// ---------- edge pass 2: denom = segment sum of exp ----------
__global__ __launch_bounds__(256)
void gat_edge_sum(const int* __restrict__ src, const int* __restrict__ dst,
                  const float* __restrict__ as, const float* __restrict__ ad,
                  const unsigned* __restrict__ emax, float* __restrict__ denom, int E) {
  int e = blockIdx.x * blockDim.x + threadIdx.x;
  if (e >= E) return;
  int s = src[e], d = dst[e];
#pragma unroll
  for (int hd = 0; hd < HEADS; ++hd) {
    float v = as[s * HEADS + hd] + ad[d * HEADS + hd];
    v = v > 0.0f ? v : NEG_SLOPE * v;
    atomicAdd(&denom[d * HEADS + hd], __expf(v - funflip(emax[d * HEADS + hd])));
  }
}

// ---------- edge pass 3: one wave per edge; lane owns 4 channels ----------
__global__ __launch_bounds__(256)
void gat_edge_scatter(const int* __restrict__ src, const int* __restrict__ dst,
                      const float* __restrict__ as, const float* __restrict__ ad,
                      const unsigned* __restrict__ emax, const float* __restrict__ denom,
                      const float* __restrict__ h, float* __restrict__ xo, int E) {
  int w = (blockIdx.x * blockDim.x + threadIdx.x) >> 5;
  int lane = threadIdx.x & 31;
  if (w >= E) return;
  int s = src[w], d = dst[w];
  int hd = lane >> 4;                       // lanes 0-15 head0, 16-31 head1
  float v = as[s * HEADS + hd] + ad[d * HEADS + hd];
  v = v > 0.0f ? v : NEG_SLOPE * v;
  float alpha = __expf(v - funflip(emax[d * HEADS + hd])) /
                (denom[d * HEADS + hd] + 1e-16f);
  const float4 hv = *(const float4*)(h + (size_t)s * DIM + lane * 4);
  float* o = xo + (size_t)d * DIM + lane * 4;
  atomicAdd(o + 0, hv.x * alpha);
  atomicAdd(o + 1, hv.y * alpha);
  atomicAdd(o + 2, hv.z * alpha);
  atomicAdd(o + 3, hv.w * alpha);
}

extern "C" void kernel_launch(void* const* d_in, const int* in_sizes, int n_in,
                              void* d_out, int out_size, void* d_ws, size_t ws_size,
                              hipStream_t stream) {
  (void)in_sizes; (void)n_in; (void)out_size; (void)ws_size;
  const float* seq     = (const float*)d_in[0];        // [1,N,128]
  const int*   adj     = (const int*)d_in[1];          // [2,E]
  const float* W       = (const float*)d_in[2];        // [3,128,128]
  const float* att_src = (const float*)d_in[3];        // [3,2,64]
  const float* att_dst = (const float*)d_in[4];        // [3,2,64]
  const float* bias    = (const float*)d_in[5];        // [3,128]

  const int N = N_NODES, E = N_EDGES;
  const int* src = adj;
  const int* dst = adj + E;

  char* ws = (char*)d_ws;
  size_t off = 0;
  _Float16* Wt   = (_Float16*)(ws + off); off += (size_t)3 * DIM * DIM * sizeof(_Float16); // 96 KB
  float*    h    = (float*)(ws + off);    off += (size_t)N * DIM * sizeof(float);          // 51.2 MB
  float*    as   = (float*)(ws + off);    off += (size_t)N * HEADS * sizeof(float);
  float*    ad   = (float*)(ws + off);    off += (size_t)N * HEADS * sizeof(float);
  unsigned* emax = (unsigned*)(ws + off); off += (size_t)N * HEADS * sizeof(unsigned);
  float*    den  = (float*)(ws + off);    off += (size_t)N * HEADS * sizeof(float);
  float*    bufA = (float*)(ws + off);    off += (size_t)N * DIM * sizeof(float);
  float*    bufB = (float*)(ws + off);    off += (size_t)N * DIM * sizeof(float);

  const int convTot = 3 * DIM * DIM;
  gat_convW<<<(convTot + 255) / 256, 256, 0, stream>>>(W, Wt, convTot);

  const float* xin = seq;
  for (int l = 0; l < 3; ++l) {
    float* xo = (l == 0) ? bufA : (l == 1) ? bufB : (float*)d_out;

    gat_gemm<<<N / 16, 256, 0, stream>>>(xin, Wt + (size_t)l * DIM * DIM, h);
    gat_alpha<<<(N + 255) / 256, 256, 0, stream>>>(h, att_src + l * HEADS * CH,
                                                   att_dst + l * HEADS * CH, as, ad, N);
    gat_init<<<((size_t)N * DIM + 255) / 256, 256, 0, stream>>>(
        xo, l ? xin : nullptr, bias + l * DIM, emax, den, N * DIM);
    gat_edge_max<<<(E + 255) / 256, 256, 0, stream>>>(src, dst, as, ad, emax, E);
    gat_edge_sum<<<(E + 255) / 256, 256, 0, stream>>>(src, dst, as, ad, emax, den, E);
    gat_edge_scatter<<<((size_t)E * 32 + 255) / 256, 256, 0, stream>>>(
        src, dst, as, ad, emax, den, h, xo, E);

    xin = xo;
  }
}
